// LCAWSupConLoss_55052890800865
// MI455X (gfx1250) — matrix operand order repaired and checked by reading, not compiled, not run on hardware
//
#include <hip/hip_runtime.h>
#include <hip/hip_bf16.h>
#include <math.h>

#define N_ROWS 8192
#define DIM    1024
#define TILE   16
#define ROWS_PER_WG 32                 // two 16-row WMMA tiles per workgroup
#define NTILES (N_ROWS / TILE)         // 512 column tiles
#define WAVES  8
#define KCH    (DIM / 32)              // 32 K-chunks of 32

typedef __attribute__((ext_vector_type(16))) _Float16 v16h;
typedef __attribute__((ext_vector_type(8)))  _Float16 v8h;
typedef __attribute__((ext_vector_type(8)))  float    v8f;

union V16U { v16h v; v8h h[2]; };

__global__ void zero_out_kernel(float* out) { out[0] = 0.0f; }

// Row L2-normalize (eps = 1e-12 like torch F.normalize) and convert to f16.
__global__ __launch_bounds__(256) void normalize_f16_kernel(const float* __restrict__ F,
                                                            _Float16* __restrict__ Fh) {
  __shared__ float red[256];
  const int row = blockIdx.x;
  const int tid = threadIdx.x;
  const float* src = F + (size_t)row * DIM;
  float ss = 0.0f;
#pragma unroll
  for (int c = tid; c < DIM; c += 256) { float v = src[c]; ss += v * v; }
  red[tid] = ss;
  __syncthreads();
  for (int off = 128; off > 0; off >>= 1) {
    if (tid < off) red[tid] += red[tid + off];
    __syncthreads();
  }
  const float scale = 1.0f / fmaxf(sqrtf(red[0]), 1e-12f);
  _Float16* dst = Fh + (size_t)row * DIM;
#pragma unroll
  for (int c = tid; c < DIM; c += 256) dst[c] = (_Float16)(src[c] * scale);
}

// Each workgroup owns 32 rows (2 WMMA row-tiles in LDS). 8 waves split the 512
// column tiles; each wave streams B fragments from L2 and reuses each B chunk
// for two v_wmma_f32_16x16x32_f16 (one per row-tile). Row statistics
// (sum exp(sim-1), sum of positive sims, positive count) accumulate in VGPRs.
// Row labels are hoisted into registers so the per-tile epilogue is branch-free
// pure VALU (co-executes with the XDL WMMA pipeline).
__global__ __launch_bounds__(256) void supcon_kernel(const _Float16* __restrict__ Fh,
                                                     const int* __restrict__ labels,
                                                     float* __restrict__ out) {
  __shared__ _Float16 Atile[ROWS_PER_WG * DIM];      // 64 KB
  __shared__ float parts[WAVES][ROWS_PER_WG][3];

  const int tid  = threadIdx.x;
  const int wave = tid >> 5;
  const int lane = tid & 31;
  const int half = lane >> 4;      // 0: lanes 0-15, 1: lanes 16-31
  const int lmod = lane & 15;
  const int row_base = blockIdx.x * ROWS_PER_WG;

  // Stage the 32 owned rows into LDS (16-byte chunks).
  {
    const int4* g = (const int4*)(Fh + (size_t)row_base * DIM);
    int4* s = (int4*)Atile;
    for (int i = tid; i < ROWS_PER_WG * DIM * 2 / 16; i += 256) s[i] = g[i];
  }
  __syncthreads();

  // Register-resident row labels and global row ids for the 16 (t,r) slots this
  // lane owns: m = t*16 + half*8 + r  (C/D layout: half 0 -> M=r, half 1 -> M=r+8).
  int rl[2][8];
  int gi[2][8];
#pragma unroll
  for (int t = 0; t < 2; ++t)
#pragma unroll
    for (int r = 0; r < 8; ++r) {
      const int m = t * TILE + half * 8 + r;
      gi[t][r] = row_base + m;
      rl[t][r] = labels[row_base + m];
    }

  // Per-lane statistics: [row-tile][accumulator slot r]
  float sumexp[2][8], possum[2][8], npos[2][8];
#pragma unroll
  for (int t = 0; t < 2; ++t)
#pragma unroll
    for (int r = 0; r < 8; ++r) { sumexp[t][r] = 0.f; possum[t][r] = 0.f; npos[t][r] = 0.f; }

  // A-fragment base (ISA 16-bit A layout: lane<16 -> K {0..7,16..23},
  // lane>=16 -> K {8..15,24..31}, row M = lane&15).
  const _Float16* a_base0 = Atile + (size_t)lmod * DIM + half * 8;
  const _Float16* a_base1 = a_base0 + (size_t)TILE * DIM;

  for (int jt = wave; jt < NTILES; jt += WAVES) {
    const int col_base = jt * TILE;
    const int gj = col_base + lmod;          // global column this lane owns
    const int lab_n = labels[gj];
    // B layout: lane<16 -> K {0..7,16..23}, lane>=16 -> K {8..15,24..31} of column gj; column of f^T is a row of f.
    const _Float16* b_base = Fh + (size_t)gj * DIM + half * 8;

    v8f acc0 = {};
    v8f acc1 = {};
#pragma unroll 4
    for (int kc = 0; kc < KCH; ++kc) {
      const int k0 = kc * 32;
      V16U a0, a1, b;
      a0.h[0] = *(const v8h*)(a_base0 + k0);
      a0.h[1] = *(const v8h*)(a_base0 + k0 + 16);
      a1.h[0] = *(const v8h*)(a_base1 + k0);
      a1.h[1] = *(const v8h*)(a_base1 + k0 + 16);
      b.h[0]  = *(const v8h*)(b_base + k0);
      b.h[1]  = *(const v8h*)(b_base + k0 + 16);
      acc0 = __builtin_amdgcn_wmma_f32_16x16x32_f16(false, a0.v, false, b.v,
                                                    (short)0, acc0, false, false);
      acc1 = __builtin_amdgcn_wmma_f32_16x16x32_f16(false, a1.v, false, b.v,
                                                    (short)0, acc1, false, false);
    }

    // Branch-free statistics update (pure VALU: cndmask + v_exp_f32).
#pragma unroll
    for (int t = 0; t < 2; ++t) {
#pragma unroll
      for (int r = 0; r < 8; ++r) {
        const float s = (t == 0) ? acc0[r] : acc1[r];
        const bool self = (gi[t][r] == gj);
        const bool pos  = (lab_n == rl[t][r]) && !self;
        const float e = __expf(s - 1.0f);           // shift by max (cos <= 1)
        sumexp[t][r] += self ? 0.0f : e;
        possum[t][r] += pos ? s : 0.0f;
        npos[t][r]   += pos ? 1.0f : 0.0f;
      }
    }
  }

  // Reduce over the 16 lanes (columns) of each half-wave.
#pragma unroll
  for (int t = 0; t < 2; ++t)
#pragma unroll
    for (int r = 0; r < 8; ++r) {
#pragma unroll
      for (int m = 1; m <= 8; m <<= 1) {
        sumexp[t][r] += __shfl_xor(sumexp[t][r], m, 16);
        possum[t][r] += __shfl_xor(possum[t][r], m, 16);
        npos[t][r]   += __shfl_xor(npos[t][r],   m, 16);
      }
    }
  if (lmod == 0) {
#pragma unroll
    for (int t = 0; t < 2; ++t)
#pragma unroll
      for (int r = 0; r < 8; ++r) {
        const int m = t * TILE + half * 8 + r;
        parts[wave][m][0] = sumexp[t][r];
        parts[wave][m][1] = possum[t][r];
        parts[wave][m][2] = npos[t][r];
      }
  }
  __syncthreads();

  // Combine wave partials, finish each row, accumulate scalar loss.
  if (tid < ROWS_PER_WG) {
    float se = 0.f, ps = 0.f, np = 0.f;
#pragma unroll
    for (int w = 0; w < WAVES; ++w) {
      se += parts[w][tid][0];
      ps += parts[w][tid][1];
      np += parts[w][tid][2];
    }
    const float lse = 1.0f + __logf(se);          // undo the exp(s-1) shift
    const float li  = -(ps - np * lse) / (np + 1e-5f);
    atomicAdd(out, li);
  }
}

extern "C" void kernel_launch(void* const* d_in, const int* in_sizes, int n_in,
                              void* d_out, int out_size, void* d_ws, size_t ws_size,
                              hipStream_t stream) {
  const float* features = (const float*)d_in[0];
  const int*   labels   = (const int*)d_in[1];
  float* out = (float*)d_out;
  _Float16* Fh = (_Float16*)d_ws;                 // 8192*1024*2 = 16 MB scratch

  zero_out_kernel<<<1, 1, 0, stream>>>(out);
  normalize_f16_kernel<<<N_ROWS, 256, 0, stream>>>(features, Fh);
  supcon_kernel<<<N_ROWS / ROWS_PER_WG, 256, 0, stream>>>(Fh, labels, out);
}
